// Bert_69698729280007
// MI455X (gfx1250) — matrix operand order repaired and checked by reading, not compiled, not run on hardware
//
#include <hip/hip_runtime.h>

typedef __attribute__((ext_vector_type(16))) _Float16 v16h;
typedef __attribute__((ext_vector_type(8)))  float    v8f;
typedef _Float16 f16;

#define NSEQ   16
#define CHNK   512
#define DMODEL 768
#define NHEAD  12
#define DHEAD  64
#define FFDIM  3072
#define NLAYER 4
#define MTOT   (NSEQ*CHNK)   /* 8192 */
#define BATCH  8
#define SEQLEN 1024
#define NWORD  512
#define QKVN   (3*DMODEL)    /* 2304 packed QKV width */
#define ATTN_LDS (64*512*4 + 64*512*2 + 64*512*2)  /* 256 KB: sS(f32)+sP(f16)+sVt(f16) */

// ---------------------------------------------------------------------------
// CDNA5 primitives
// ---------------------------------------------------------------------------
__device__ __forceinline__ v8f wmma16(v16h a, v16h b, v8f c) {
  return __builtin_amdgcn_wmma_f32_16x16x32_f16(false, a, false, b, (short)0, c,
                                                false, false);
}

// Async copy 16B/lane from global to LDS (ASYNCcnt-tracked, no VGPR data).
__device__ __forceinline__ void async_ld_b128(const f16* gptr, const f16* lptr) {
  const unsigned lds = (unsigned)(unsigned long long)(const void*)lptr; // low 32b = LDS offset
  const unsigned long long ga = (unsigned long long)gptr;
  asm volatile("global_load_async_to_lds_b128 %0, %1, off"
               :: "v"(lds), "v"(ga) : "memory");
}
__device__ __forceinline__ void wait_async_le3() {
  asm volatile("s_wait_asynccnt 0x3" ::: "memory");
}
__device__ __forceinline__ void wait_async_0() {
  asm volatile("s_wait_asynccnt 0x0" ::: "memory");
}

// ---------------------------------------------------------------------------
// Generic WMMA GEMM:  out = act( A[MxK](f16) @ Wt[NxK]^T(f16) + bias + res )
// 256 threads (8 waves), tile 64(M) x 128(N); wave tile 32x32 (2x2 WMMA).
// Double-buffered async-to-LDS pipeline over K (32 per step).
// ---------------------------------------------------------------------------
__global__ void __launch_bounds__(256)
wmma_gemm(const f16* __restrict__ A, const f16* __restrict__ Wt,
          const float* __restrict__ bias, const float* __restrict__ res,
          float* __restrict__ out32, f16* __restrict__ out16,
          int M, int N, int K, int gelu)
{
  __shared__ f16 sA[2][64*32];    // 2 x 4 KB
  __shared__ f16 sB[2][128*32];   // 2 x 8 KB
  const int tid  = threadIdx.x;
  const int wave = tid >> 5, lane = tid & 31;
  const int wm = wave >> 2, wn = wave & 3;            // 2 x 4 wave grid
  const int bm = blockIdx.y * 64, bn = blockIdx.x * 128;
  const int row = lane & 15, kh = lane >> 4;

  const v8f zero8 = {0.f,0.f,0.f,0.f,0.f,0.f,0.f,0.f};
  v8f acc[2][2];
  acc[0][0]=zero8; acc[0][1]=zero8; acc[1][0]=zero8; acc[1][1]=zero8;

  // per-thread staging chunks: 1 for A (64x32), 2 for B (128x32)
  const int ra = tid >> 2, sa = (tid & 3) * 8;
  const int nb = tid >> 2, sb = (tid & 3) * 8;        // B rows nb and nb+64

  const f16* gA  = A  + (size_t)(bm + ra)*K + sa;
  const f16* gB0 = Wt + (size_t)(bn + nb)*K + sb;
  const f16* gB1 = Wt + (size_t)(bn + nb + 64)*K + sb;

  // prologue: tile 0 -> buffer 0
  async_ld_b128(gA,        &sA[0][ra*32 + sa]);
  async_ld_b128(gB0,       &sB[0][nb*32 + sb]);
  async_ld_b128(gB1,       &sB[0][(nb+64)*32 + sb]);

  int buf = 0;
  for (int k0 = 0; k0 < K; k0 += 32, buf ^= 1) {
    if (k0 + 32 < K) {      // issue next tile into the other buffer
      const int nx = buf ^ 1, kn = k0 + 32;
      async_ld_b128(gA  + kn, &sA[nx][ra*32 + sa]);
      async_ld_b128(gB0 + kn, &sB[nx][nb*32 + sb]);
      async_ld_b128(gB1 + kn, &sB[nx][(nb+64)*32 + sb]);
      wait_async_le3();     // in-order completion => current tile has landed
    } else {
      wait_async_0();
    }
    __syncthreads();        // all waves' tile data visible in LDS

    v16h a[2], b[2];
#pragma unroll
    for (int i = 0; i < 2; ++i) {                 // A frag: K runs of 8 at kh*8, kh*8+16
      const f16* p = &sA[buf][(wm*32 + i*16 + row)*32 + kh*8];
      ((uint4*)&a[i])[0] = *(const uint4*)p;
      ((uint4*)&a[i])[1] = *(const uint4*)(p + 16);
    }
#pragma unroll
    for (int j = 0; j < 2; ++j) {                 // B frag: K runs of 8 at kh*8, kh*8+16
      const f16* p = &sB[buf][(wn*32 + j*16 + row)*32 + kh*8];
      ((uint4*)&b[j])[0] = *(const uint4*)p;
      ((uint4*)&b[j])[1] = *(const uint4*)(p + 16);
    }
#pragma unroll
    for (int i = 0; i < 2; ++i)
#pragma unroll
      for (int j = 0; j < 2; ++j)
        acc[i][j] = wmma16(a[i], b[j], acc[i][j]);

    __syncthreads();        // done reading this buffer before it is re-filled
  }

  // epilogue: C/D layout -> lane N = lane&15, M = 8*(lane>>4)+r
#pragma unroll
  for (int i = 0; i < 2; ++i)
#pragma unroll
    for (int j = 0; j < 2; ++j) {
      const int gN = bn + wn*32 + j*16 + row;
      const float bv = bias ? bias[gN] : 0.f;
#pragma unroll
      for (int r = 0; r < 8; ++r) {
        const int gM = bm + wm*32 + i*16 + kh*8 + r;
        float v = acc[i][j][r] + bv;
        if (res)   v += res[(size_t)gM*N + gN];
        if (gelu)  v = 0.5f * v * (1.f + erff(v * 0.70710678118f));
        if (out32) out32[(size_t)gM*N + gN] = v;
        if (out16) out16[(size_t)gM*N + gN] = (f16)v;
      }
    }
}

// ---------------------------------------------------------------------------
// Fused attention on packed QKV [MTOT, 2304]: per (64-query block, head, seq)
// S = Q K^T * scale + mask-bias ; softmax ; ctx = P V   (both matmuls on WMMA)
// ---------------------------------------------------------------------------
__global__ void __launch_bounds__(256)
attn_kernel(const f16* __restrict__ qkv, const float* __restrict__ mask,
            f16* __restrict__ ctx16)
{
  extern __shared__ char smem[];
  float* sS  = (float*)smem;                         // 64 x 512 f32 scores
  f16*   sP  = (f16*)(smem + 64*512*4);              // 64 x 512 f16 probs
  f16*   sVt = (f16*)(smem + 64*512*4 + 64*512*2);   // 64(dh) x 512(key) f16 V^T
  const int qb = blockIdx.x, hd = blockIdx.y, s = blockIdx.z;
  const int tid = threadIdx.x, wave = tid >> 5, lane = tid & 31;
  const int wm = wave >> 2, wn = wave & 3;
  const int row = lane & 15, kh = lane >> 4;
  const v8f zero8 = {0.f,0.f,0.f,0.f,0.f,0.f,0.f,0.f};

  const f16* qbase = qkv;                 // col 0..767
  const f16* kbase = qkv + DMODEL;        // col 768..1535
  const f16* vbase = qkv + 2*DMODEL;      // col 1536..2303

  // stage V^T (dh-major) so the P@V B-fragment reads are contiguous
  for (int c = tid; c < 4096; c += 256) {
    const int key = c >> 3, dseg = (c & 7) * 8;
    uint4 u = *(const uint4*)(vbase + ((size_t)s*CHNK + key)*QKVN + hd*DHEAD + dseg);
    const f16* t = (const f16*)&u;
#pragma unroll
    for (int i = 0; i < 8; ++i) sVt[(dseg + i)*CHNK + key] = t[i];
  }
  __syncthreads();

  // ---- scores: wave covers 32 queries x 128 keys, in 32-key subtiles ----
  const size_t qrow0 = (size_t)s*CHNK + qb*64 + wm*32;
  for (int nt = 0; nt < 4; ++nt) {
    const int key0 = wn*128 + nt*32;
    v8f acc[2][2];
    acc[0][0]=zero8; acc[0][1]=zero8; acc[1][0]=zero8; acc[1][1]=zero8;
#pragma unroll
    for (int ks = 0; ks < DHEAD; ks += 32) {
      v16h a[2], b[2];
#pragma unroll
      for (int i = 0; i < 2; ++i) {
        const f16* p = qbase + (qrow0 + i*16 + row)*QKVN + hd*DHEAD + ks + kh*8;
        ((uint4*)&a[i])[0] = *(const uint4*)p;
        ((uint4*)&a[i])[1] = *(const uint4*)(p + 16);
      }
#pragma unroll
      for (int j = 0; j < 2; ++j) {   // K matrix acts as NxK for the B fragment
        const f16* p = kbase + ((size_t)s*CHNK + key0 + j*16 + row)*QKVN + hd*DHEAD + ks + kh*8;
        ((uint4*)&b[j])[0] = *(const uint4*)p;
        ((uint4*)&b[j])[1] = *(const uint4*)(p + 16);
      }
#pragma unroll
      for (int i = 0; i < 2; ++i)
#pragma unroll
        for (int j = 0; j < 2; ++j) acc[i][j] = wmma16(a[i], b[j], acc[i][j]);
    }
#pragma unroll
    for (int i = 0; i < 2; ++i)
#pragma unroll
      for (int j = 0; j < 2; ++j) {
        const int key = key0 + j*16 + row;
        const float mb = (1.0f - mask[s*CHNK + key]) * -10000.0f;
#pragma unroll
        for (int r = 0; r < 8; ++r) {
          const int lq = wm*32 + i*16 + kh*8 + r;
          sS[lq*CHNK + key] = acc[i][j][r] * 0.125f + mb;  // 1/sqrt(64)
        }
      }
  }
  __syncthreads();

  // ---- softmax: one wave per row (8 rows/wave) ----
  for (int rr = wave; rr < 64; rr += 8) {
    float* xr = sS + rr*CHNK;
    float mx = -3.0e38f;
    for (int i = lane; i < CHNK; i += 32) mx = fmaxf(mx, xr[i]);
#pragma unroll
    for (int o = 16; o; o >>= 1) mx = fmaxf(mx, __shfl_xor(mx, o, 32));
    float sm = 0.f;
    for (int i = lane; i < CHNK; i += 32) sm += expf(xr[i] - mx);
#pragma unroll
    for (int o = 16; o; o >>= 1) sm += __shfl_xor(sm, o, 32);
    const float inv = 1.f / sm;
    for (int i = lane; i < CHNK; i += 32) sP[rr*CHNK + i] = (f16)(expf(xr[i] - mx) * inv);
  }
  __syncthreads();

  // ---- ctx = P(64x512) @ V(512x64): wave covers 32 q x 16 dh ----
  v8f c2[2]; c2[0] = zero8; c2[1] = zero8;
  for (int kk = 0; kk < CHNK; kk += 32) {
    v16h a[2], b;
#pragma unroll
    for (int i = 0; i < 2; ++i) {
      const f16* p = sP + (wm*32 + i*16 + row)*CHNK + kk + kh*8;
      ((uint4*)&a[i])[0] = *(const uint4*)p;
      ((uint4*)&a[i])[1] = *(const uint4*)(p + 16);
    }
    {
      const f16* p = sVt + (wn*16 + row)*CHNK + kk + kh*8;
      ((uint4*)&b)[0] = *(const uint4*)p;
      ((uint4*)&b)[1] = *(const uint4*)(p + 16);
    }
#pragma unroll
    for (int i = 0; i < 2; ++i) c2[i] = wmma16(a[i], b, c2[i]);
  }
#pragma unroll
  for (int i = 0; i < 2; ++i)
#pragma unroll
    for (int r = 0; r < 8; ++r) {
      const int gq = s*CHNK + qb*64 + wm*32 + i*16 + kh*8 + r;
      const int n  = wn*16 + row;
      ctx16[(size_t)gq*DMODEL + hd*DHEAD + n] = (f16)c2[i][r];
    }
}

// ---------------------------------------------------------------------------
// Embedding gather: out = word_emb[id] + pos_emb[pos] + type_emb[0]
// ---------------------------------------------------------------------------
__global__ void __launch_bounds__(256)
embed_kernel(const int* __restrict__ ids, const float* __restrict__ we,
             const float* __restrict__ pe, const float* __restrict__ te,
             float* __restrict__ out)
{
  const int row = blockIdx.x;          // 0..8191 over [16,512]
  const int tok = ids[row];
  const int p   = row & (CHNK - 1);
  const float* wr = we + (size_t)tok*DMODEL;
  const float* pr = pe + (size_t)p*DMODEL;
  float* o = out + (size_t)row*DMODEL;
  for (int d = threadIdx.x; d < DMODEL; d += 256) o[d] = wr[d] + pr[d] + te[d];
}

// ---------------------------------------------------------------------------
// LayerNorm over rows of 768; writes f32 master + f16 GEMM input
// ---------------------------------------------------------------------------
__device__ __forceinline__ float block_sum(float v, float* sbuf) {
#pragma unroll
  for (int o = 16; o; o >>= 1) v += __shfl_xor(v, o, 32);
  const int wv = threadIdx.x >> 5, ln = threadIdx.x & 31;
  if (ln == 0) sbuf[wv] = v;
  __syncthreads();
  float t = sbuf[0];
#pragma unroll
  for (int i = 1; i < 8; ++i) t += sbuf[i];
  __syncthreads();
  return t;
}

__global__ void __launch_bounds__(256)
ln_rows(const float* __restrict__ in, const float* __restrict__ g,
        const float* __restrict__ b, float* __restrict__ out32,
        f16* __restrict__ out16)
{
  __shared__ float sbuf[8];
  const int row = blockIdx.x, tid = threadIdx.x;
  const float* x = in + (size_t)row*DMODEL;
  const float v0 = x[tid], v1 = x[tid + 256], v2 = x[tid + 512];
  const float mu = block_sum(v0 + v1 + v2, sbuf) * (1.f/DMODEL);
  const float d0 = v0 - mu, d1 = v1 - mu, d2 = v2 - mu;
  const float var = block_sum(d0*d0 + d1*d1 + d2*d2, sbuf) * (1.f/DMODEL);
  const float inv = rsqrtf(var + 1e-12f);
  const float o0 = d0*inv*g[tid]       + b[tid];
  const float o1 = d1*inv*g[tid + 256] + b[tid + 256];
  const float o2 = d2*inv*g[tid + 512] + b[tid + 512];
  float* o  = out32 + (size_t)row*DMODEL;
  f16*   oh = out16 + (size_t)row*DMODEL;
  o[tid] = o0; o[tid + 256] = o1; o[tid + 512] = o2;
  oh[tid] = (f16)o0; oh[tid + 256] = (f16)o1; oh[tid + 512] = (f16)o2;
}

// ---------------------------------------------------------------------------
// Weight convert+transpose: Wsrc[K,N] f32 -> Wt[N,K] f16 (Wt may be offset)
// ---------------------------------------------------------------------------
__global__ void __launch_bounds__(256)
convert_wt(const float* __restrict__ Wsrc, f16* __restrict__ Wt, int K, int N)
{
  const size_t idx = (size_t)blockIdx.x*256 + threadIdx.x;
  const int n = (int)(idx / K), k = (int)(idx % K);
  Wt[idx] = (f16)Wsrc[(size_t)k*N + n];
}

// pack three 768-float bias vectors into one 2304 vector
__global__ void __launch_bounds__(256)
pack_bias3(const float* __restrict__ a, const float* __restrict__ b,
           const float* __restrict__ c, float* __restrict__ o)
{
  const int i = blockIdx.x*256 + threadIdx.x;  // < 2304
  o[i] = (i < DMODEL) ? a[i] : (i < 2*DMODEL ? b[i - DMODEL] : c[i - 2*DMODEL]);
}

// ---------------------------------------------------------------------------
// Offset-based mean pooling over word spans
// ---------------------------------------------------------------------------
__global__ void __launch_bounds__(256)
pool_kernel(const float* __restrict__ emb, const int* __restrict__ off,
            const int* __restrict__ msk, float* __restrict__ out)
{
  const int idx = blockIdx.x;              // b*512 + w
  const int b = idx >> 9;
  const int st = off[idx*2], ed = off[idx*2 + 1];
  const bool valid = (msk[idx] != 0) && (st < ed);
  const float inv = valid ? 1.0f / (float)(ed - st) : 0.f;
  for (int d = threadIdx.x; d < DMODEL; d += 256) {
    float s = 0.f;
    if (valid)
      for (int t = st; t < ed; ++t)
        s += emb[((size_t)b*SEQLEN + t)*DMODEL + d];
    out[(size_t)idx*DMODEL + d] = s * inv;
  }
}

// ---------------------------------------------------------------------------
// Host orchestration
// ---------------------------------------------------------------------------
extern "C" void kernel_launch(void* const* d_in, const int* in_sizes, int n_in,
                              void* d_out, int out_size, void* d_ws, size_t ws_size,
                              hipStream_t stream)
{
  const int*   x_bert  = (const int*)  d_in[0];
  const float* x_bmask = (const float*)d_in[1];
  const int*   x_off   = (const int*)  d_in[2];
  const int*   x_mask  = (const int*)  d_in[3];
  const float* word_e  = (const float*)d_in[4];
  const float* pos_e   = (const float*)d_in[5];
  const float* type_e  = (const float*)d_in[6];
  const float* emb_g   = (const float*)d_in[7];
  const float* emb_b   = (const float*)d_in[8];
  const float* Wq = (const float*)d_in[9];  const float* bq = (const float*)d_in[10];
  const float* Wk = (const float*)d_in[11]; const float* bk = (const float*)d_in[12];
  const float* Wv = (const float*)d_in[13]; const float* bv = (const float*)d_in[14];
  const float* Wo = (const float*)d_in[15]; const float* bo = (const float*)d_in[16];
  const float* g1 = (const float*)d_in[17]; const float* be1 = (const float*)d_in[18];
  const float* W1 = (const float*)d_in[19]; const float* b1 = (const float*)d_in[20];
  const float* W2 = (const float*)d_in[21]; const float* b2 = (const float*)d_in[22];
  const float* g2 = (const float*)d_in[23]; const float* be2 = (const float*)d_in[24];

  // workspace carve
  char* wp = (char*)d_ws;
  auto take = [&](size_t bytes) { char* p = wp; wp += (bytes + 255) & ~(size_t)255; return p; };
  float* h32   = (float*)take((size_t)MTOT*DMODEL*4);
  float* y32   = (float*)take((size_t)MTOT*DMODEL*4);
  f16*   h16   = (f16*)  take((size_t)MTOT*DMODEL*2);
  f16*   qkv16 = (f16*)  take((size_t)MTOT*QKVN*2);
  f16*   c16   = (f16*)  take((size_t)MTOT*DMODEL*2);
  f16*   f116  = (f16*)  take((size_t)MTOT*FFDIM*2);
  f16*   wt16  = (f16*)  take((size_t)DMODEL*FFDIM*2);
  float* b3    = (float*)take((size_t)QKVN*4);

  hipFuncSetAttribute(reinterpret_cast<const void*>(attn_kernel),
                      hipFuncAttributeMaxDynamicSharedMemorySize, ATTN_LDS);

  // embedding + LN
  embed_kernel<<<MTOT, 256, 0, stream>>>(x_bert, word_e, pos_e, type_e, y32);
  ln_rows<<<MTOT, 256, 0, stream>>>(y32, emb_g, emb_b, h32, h16);

  const dim3 gQKV(QKVN/128,  MTOT/64);  // N=2304
  const dim3 gD  (DMODEL/128, MTOT/64); // N=768
  const dim3 gF  (FFDIM/128,  MTOT/64); // N=3072
  for (int l = 0; l < NLAYER; ++l) {
    const float* wq = Wq + (size_t)l*DMODEL*DMODEL; const float* bq_ = bq + l*DMODEL;
    const float* wk = Wk + (size_t)l*DMODEL*DMODEL; const float* bk_ = bk + l*DMODEL;
    const float* wv = Wv + (size_t)l*DMODEL*DMODEL; const float* bv_ = bv + l*DMODEL;
    const float* wo = Wo + (size_t)l*DMODEL*DMODEL; const float* bo_ = bo + l*DMODEL;
    const float* w1 = W1 + (size_t)l*DMODEL*FFDIM;  const float* b1_ = b1 + l*FFDIM;
    const float* w2 = W2 + (size_t)l*FFDIM*DMODEL;  const float* b2_ = b2 + l*DMODEL;
    const float* g1_ = g1 + l*DMODEL; const float* be1_ = be1 + l*DMODEL;
    const float* g2_ = g2 + l*DMODEL; const float* be2_ = be2 + l*DMODEL;

    // packed QKV projection: Wt rows [0,768)=Q, [768,1536)=K, [1536,2304)=V
    convert_wt<<<(DMODEL*DMODEL)/256, 256, 0, stream>>>(wq, wt16,                     DMODEL, DMODEL);
    convert_wt<<<(DMODEL*DMODEL)/256, 256, 0, stream>>>(wk, wt16 +   DMODEL*DMODEL,   DMODEL, DMODEL);
    convert_wt<<<(DMODEL*DMODEL)/256, 256, 0, stream>>>(wv, wt16 + 2*DMODEL*DMODEL,   DMODEL, DMODEL);
    pack_bias3<<<QKVN/256, 256, 0, stream>>>(bq_, bk_, bv_, b3);
    wmma_gemm<<<gQKV, 256, 0, stream>>>(h16, wt16, b3, nullptr, nullptr, qkv16,
                                        MTOT, QKVN, DMODEL, 0);

    // attention
    attn_kernel<<<dim3(CHNK/64, NHEAD, NSEQ), 256, ATTN_LDS, stream>>>(
        qkv16, x_bmask, c16);

    // output projection + residual, then LN1
    convert_wt<<<(DMODEL*DMODEL)/256, 256, 0, stream>>>(wo, wt16, DMODEL, DMODEL);
    wmma_gemm<<<gD, 256, 0, stream>>>(c16, wt16, bo_, h32, y32, nullptr,
                                      MTOT, DMODEL, DMODEL, 0);
    ln_rows<<<MTOT, 256, 0, stream>>>(y32, g1_, be1_, h32, h16);

    // FFN1 with fused exact GELU (f16 output)
    convert_wt<<<(DMODEL*FFDIM)/256, 256, 0, stream>>>(w1, wt16, DMODEL, FFDIM);
    wmma_gemm<<<gF, 256, 0, stream>>>(h16, wt16, b1_, nullptr, nullptr, f116,
                                      MTOT, FFDIM, DMODEL, 1);
    // FFN2 + residual, then LN2
    convert_wt<<<(DMODEL*FFDIM)/256, 256, 0, stream>>>(w2, wt16, FFDIM, DMODEL);
    wmma_gemm<<<gD, 256, 0, stream>>>(f116, wt16, b2_, h32, y32, nullptr,
                                      MTOT, DMODEL, FFDIM, 0);
    ln_rows<<<MTOT, 256, 0, stream>>>(y32, g2_, be2_, h32, h16);
  }

  // offset mean pooling -> d_out [8,512,768] f32
  pool_kernel<<<BATCH*NWORD, 256, 0, stream>>>(h32, x_off, x_mask, (float*)d_out);
}
